// QuantumAttention_43439299232071
// MI455X (gfx1250) — matrix operand order repaired and checked by reading, not compiled, not run on hardware
//
#include <hip/hip_runtime.h>
#include <hip/hip_bf16.h>

// ---------------------------------------------------------------------------
// QuantumAttention on MI455X (gfx1250, wave32):
//   - bf16 WMMA 16x16x32 for all three large GEMMs
//   - async global->LDS (ASYNCcnt) double-buffered staging, no in-kernel
//     transpose (all B operands pre-transposed in global memory)
//   - algebraic fusion kills the full Q/K projections (only q@We survives)
// ---------------------------------------------------------------------------

#define HID   1024
#define QDIM  16
#define BATCH 4
#define SEQ   2048
#define ROWS  (BATCH * SEQ)          // 8192
#define EPSQ  1e-8f

#define USE_ASYNC_TLDS 1             // gfx1250 global_load_async_to_lds path

typedef __attribute__((ext_vector_type(16))) __bf16 v16bf;
typedef __attribute__((ext_vector_type(8)))  float  v8f;

union AFrag { unsigned int u[8]; v16bf v; };
static_assert(sizeof(AFrag) == 32, "frag size");

__device__ __forceinline__ unsigned short f2bf(float f) {
    unsigned int u = __builtin_bit_cast(unsigned int, f);
    u += 0x7fffu + ((u >> 16) & 1u);           // round-to-nearest-even
    return (unsigned short)(u >> 16);
}

// 16-byte global -> LDS stage (async on CDNA5, sync fallback otherwise)
__device__ __forceinline__ void stage16(unsigned short* lds, const void* g) {
#if USE_ASYNC_TLDS
    asm volatile("global_load_async_to_lds_b128 %0, %1, off"
                 :: "v"((unsigned int)(unsigned long long)lds), "v"(g)
                 : "memory");
#else
    *(uint4*)lds = *(const uint4*)g;
#endif
}
__device__ __forceinline__ void stage_wait_pending3() {
#if USE_ASYNC_TLDS
    asm volatile("s_wait_asynccnt 0x3" ::: "memory");
#endif
}
__device__ __forceinline__ void stage_wait_all() {
#if USE_ASYNC_TLDS
    asm volatile("s_wait_asynccnt 0x0" ::: "memory");
#endif
}

// ---------------------------------------------------------------------------
// fp32 -> bf16 bulk convert (row-major preserved)
// ---------------------------------------------------------------------------
__global__ void cvt_bf16_kernel(const float* __restrict__ in,
                                unsigned short* __restrict__ out, int n) {
    for (int i = blockIdx.x * blockDim.x + threadIdx.x; i < n;
         i += gridDim.x * blockDim.x)
        out[i] = f2bf(in[i]);
}

// fp32 [R][C] -> bf16 transposed [C][R], 32x32 LDS tiles
__global__ void cvt_bf16_T_kernel(const float* __restrict__ in,
                                  unsigned short* __restrict__ out,
                                  int R, int C) {
    __shared__ float t[32][33];
    const int r0 = blockIdx.y * 32, c0 = blockIdx.x * 32;
    const int tx = threadIdx.x & 31, ty = threadIdx.x >> 5;   // 256 thr: ty 0..7
    #pragma unroll
    for (int i = ty; i < 32; i += 8)
        t[i][tx] = in[(size_t)(r0 + i) * C + c0 + tx];
    __syncthreads();
    #pragma unroll
    for (int i = ty; i < 32; i += 8)
        out[(size_t)(c0 + i) * R + r0 + tx] = f2bf(t[tx][i]);
}

// ---------------------------------------------------------------------------
// Wqe = Wq @ We, Wke = Wk @ We  (thin [HIDx16] folds; VALU-bound, negligible)
// ---------------------------------------------------------------------------
__global__ void folded_we_kernel(const float* __restrict__ Wq,
                                 const float* __restrict__ Wk,
                                 const float* __restrict__ We,
                                 float* __restrict__ Wqe,
                                 float* __restrict__ Wke) {
    int id = blockIdx.x * blockDim.x + threadIdx.x;
    if (id >= 2 * HID * QDIM) return;
    const float* W = (id < HID * QDIM) ? Wq : Wk;
    float*       O = (id < HID * QDIM) ? Wqe : Wke;
    int lid = id % (HID * QDIM);
    int i = lid >> 4, j = lid & 15;
    float acc = 0.f;
    const float* wr = W + (size_t)i * HID;
    for (int m = 0; m < HID; ++m) acc += wr[m] * We[m * QDIM + j];
    O[i * QDIM + j] = acc;
}

__global__ void folded_bias_kernel(const float* __restrict__ bq,
                                   const float* __restrict__ bk,
                                   const float* __restrict__ We,
                                   const float* __restrict__ be,
                                   float* __restrict__ bqe,
                                   float* __restrict__ bke) {
    int tid = threadIdx.x;            // 32 threads
    if (tid >= 32) return;
    const float* b = (tid < 16) ? bq : bk;
    float*       o = (tid < 16) ? bqe : bke;
    int j = tid & 15;
    float acc = be[j];
    for (int m = 0; m < HID; ++m) acc += b[m] * We[m * QDIM + j];
    o[j] = acc;
}

// ---------------------------------------------------------------------------
// z = x @ Wqe + bqe (and k-path), unit-normalize -> qq, kq  [ROWS x 16]
// ---------------------------------------------------------------------------
__global__ void embed_norm_kernel(const float* __restrict__ x,
                                  const float* __restrict__ Wqe,
                                  const float* __restrict__ Wke,
                                  const float* __restrict__ bqe,
                                  const float* __restrict__ bke,
                                  float* __restrict__ qq,
                                  float* __restrict__ kq) {
    int row = blockIdx.x * 16 + (threadIdx.x >> 4);
    int j   = threadIdx.x & 15;
    if (row >= ROWS) return;
    const float* xr = x + (size_t)row * HID;
    float zq = bqe[j], zk = bke[j];
    for (int m = 0; m < HID; ++m) {
        float xv = xr[m];
        zq += xv * Wqe[m * QDIM + j];
        zk += xv * Wke[m * QDIM + j];
    }
    float nq = zq * zq, nk = zk * zk;
    #pragma unroll
    for (int off = 8; off >= 1; off >>= 1) {
        nq += __shfl_xor(nq, off, 16);
        nk += __shfl_xor(nk, off, 16);
    }
    qq[(size_t)row * QDIM + j] = zq * rsqrtf(nq + EPSQ);
    kq[(size_t)row * QDIM + j] = zk * rsqrtf(nk + EPSQ);
}

// ---------------------------------------------------------------------------
// P[b,row,:] = exp((q.k)^2) bf16; rowsum_inv = 1/sum.  Scores in [0,1] since
// q,k are unit vectors -> exp bounded in [1,e], no max subtraction needed.
// ---------------------------------------------------------------------------
__global__ void scores_kernel(const float* __restrict__ qq,
                              const float* __restrict__ kq,
                              unsigned short* __restrict__ P,
                              float* __restrict__ rowsum_inv) {
    int b = blockIdx.y, row = blockIdx.x, tid = threadIdx.x;
    __shared__ float qrow[QDIM];
    __shared__ float red[8];
    if (tid < QDIM) qrow[tid] = qq[((size_t)b * SEQ + row) * QDIM + tid];
    __syncthreads();
    unsigned short* Pr = P + (size_t)b * SEQ * SEQ + (size_t)row * SEQ;
    float local = 0.f;
    for (int key = tid; key < SEQ; key += 256) {
        const float* kp = kq + ((size_t)b * SEQ + key) * QDIM;
        float d = 0.f;
        #pragma unroll
        for (int j = 0; j < QDIM; ++j) d += qrow[j] * kp[j];
        float p = __expf(d * d);
        Pr[key] = f2bf(p);
        local += p;
    }
    #pragma unroll
    for (int off = 16; off >= 1; off >>= 1) local += __shfl_xor(local, off, 32);
    if ((tid & 31) == 0) red[tid >> 5] = local;
    __syncthreads();
    if (tid == 0) {
        float s = 0.f;
        #pragma unroll
        for (int i = 0; i < 8; ++i) s += red[i];
        rowsum_inv[(size_t)b * SEQ + row] = 1.0f / s;
    }
}

// ---------------------------------------------------------------------------
// Tiled bf16 WMMA GEMM:  C = rowscale .* (A[M,K] @ BT[N,K]^T + bias)
//   * B operand is pre-TRANSPOSED in global ([N][K] row-major) so both tiles
//     stage as contiguous 16B chunks via global_load_async_to_lds_b128.
//   * Double-buffered LDS; rows padded to 80B (5x16B) to spread banks.
//   * Macro tile 128x64, 8 waves x (2x2) v_wmma_f32_16x16x32_bf16.
//   * OUT_T: write C transposed ([N][M]) with packed b128 stores.
// M%128==0, N%64==0, K%32==0 at every call site.
// ---------------------------------------------------------------------------
#define LDS_STRIDE 40   // ushorts per row (80B = 5*16B)

template <bool OUT_F32, bool HAS_BIAS, bool HAS_RS, bool OUT_T>
__global__ void gemm_bf16_kernel(const unsigned short* __restrict__ A,
                                 const unsigned short* __restrict__ BT,
                                 const float* __restrict__ bias,
                                 const float* __restrict__ rowscale,
                                 void* __restrict__ C,
                                 int M, int N, int K,
                                 long strideA, long strideB, long strideC,
                                 int strideRS) {
    __shared__ __align__(16) unsigned short Atile[2][128 * LDS_STRIDE];
    __shared__ __align__(16) unsigned short Btile[2][64 * LDS_STRIDE];

    const size_t bz = blockIdx.z;
    A  += bz * strideA;
    BT += bz * strideB;
    if (HAS_RS) rowscale += bz * (size_t)strideRS;
    char* Cb = (char*)C + bz * strideC * (OUT_F32 ? 4 : 2);

    const int tid  = threadIdx.x;
    const int lane = tid & 31;
    const int w    = tid >> 5;
    const int half = lane >> 4;
    const int l16  = lane & 15;
    const int waveRow = (w >> 1) * 32;   // 0,32,64,96
    const int waveCol = (w & 1) * 32;    // 0,32
    const int blockRow = blockIdx.y * 128;
    const int blockCol = blockIdx.x * 64;

    v8f acc[2][2] = {};

    // stage one 32-deep k-slab into LDS buffer `buf` (3 async ops / thread)
    auto issueTile = [&](int k0, int buf) {
        #pragma unroll
        for (int i = 0; i < 2; ++i) {              // A: 128x32 bf16 = 512 chunks
            int id = tid + i * 256;
            int r = id >> 2, c = id & 3;
            stage16(&Atile[buf][r * LDS_STRIDE] + c * 8,
                    (const char*)A + (((size_t)(blockRow + r)) * K + k0) * 2 + c * 16);
        }
        {                                           // BT: 64x32 bf16 = 256 chunks
            int n = tid >> 2, c = tid & 3;
            stage16(&Btile[buf][n * LDS_STRIDE] + c * 8,
                    (const char*)BT + (((size_t)(blockCol + n)) * K + k0) * 2 + c * 16);
        }
    };

    const int nt = K / 32;
    issueTile(0, 0);

    for (int kt = 0; kt < nt; ++kt) {
        const int buf = kt & 1;
        if (kt + 1 < nt) { issueTile((kt + 1) * 32, buf ^ 1); stage_wait_pending3(); }
        else             { stage_wait_all(); }
        __syncthreads();

        const unsigned short* At = Atile[buf];
        const unsigned short* Bt = Btile[buf];
        AFrag afr[2], bfr[2];
        #pragma unroll
        for (int s = 0; s < 2; ++s) {
            const int m = waveRow + s * 16 + l16;
            #pragma unroll
            for (int i = 0; i < 8; ++i) {   // A frag: k = (i/4)*16 + half*8 + (i%4)*2
                int kk = ((i >> 2) << 4) + half * 8 + ((i & 3) << 1);
                afr[s].u[i] = *(const unsigned int*)&At[m * LDS_STRIDE + kk];
            }
            const int n = waveCol + s * 16 + l16;
            #pragma unroll
            for (int j = 0; j < 8; ++j) {   // B frag: k = (j/4)*16 + half*8 + (j%4)*2
                int kk = ((j >> 2) << 4) + half * 8 + ((j & 3) << 1);
                bfr[s].u[j] = *(const unsigned int*)&Bt[n * LDS_STRIDE + kk];
            }
        }
        #pragma unroll
        for (int sm = 0; sm < 2; ++sm)
            #pragma unroll
            for (int sn = 0; sn < 2; ++sn)
                acc[sm][sn] = __builtin_amdgcn_wmma_f32_16x16x32_bf16(
                    false, afr[sm].v, false, bfr[sn].v,
                    (short)0, acc[sm][sn], false, false);
        __syncthreads();
    }

    // epilogue (fully specialized; bias hoisted per column, rowscale per row)
    #pragma unroll
    for (int sm = 0; sm < 2; ++sm) {
        const int rowb = blockRow + waveRow + sm * 16 + half * 8;
        float rs[8];
        if (HAS_RS) {
            #pragma unroll
            for (int r = 0; r < 8; ++r) rs[r] = rowscale[rowb + r];
        }
        #pragma unroll
        for (int sn = 0; sn < 2; ++sn) {
            const int gcol = blockCol + waveCol + sn * 16 + l16;
            const float badd = HAS_BIAS ? bias[gcol] : 0.f;
            float vals[8];
            #pragma unroll
            for (int r = 0; r < 8; ++r) {
                float v = acc[sm][sn][r] + badd;
                if (HAS_RS) v *= rs[r];
                vals[r] = v;
            }
            if (OUT_F32) {
                float* Cf = (float*)Cb;
                #pragma unroll
                for (int r = 0; r < 8; ++r)
                    Cf[(size_t)(rowb + r) * N + gcol] = vals[r];
            } else if (OUT_T) {
                // transposed [N][M]: this lane's 8 rows are contiguous -> b128
                unsigned int pk[4];
                #pragma unroll
                for (int i = 0; i < 4; ++i)
                    pk[i] = (unsigned int)f2bf(vals[2 * i]) |
                            ((unsigned int)f2bf(vals[2 * i + 1]) << 16);
                *(uint4*)((unsigned short*)Cb + (size_t)gcol * M + rowb) =
                    make_uint4(pk[0], pk[1], pk[2], pk[3]);
            } else {
                unsigned short* Ch = (unsigned short*)Cb;
                #pragma unroll
                for (int r = 0; r < 8; ++r)
                    Ch[(size_t)(rowb + r) * N + gcol] = f2bf(vals[r]);
            }
        }
    }
}

// ---------------------------------------------------------------------------
extern "C" void kernel_launch(void* const* d_in, const int* in_sizes, int n_in,
                              void* d_out, int out_size, void* d_ws, size_t ws_size,
                              hipStream_t stream) {
    const float* x  = (const float*)d_in[0];
    const float* Wq = (const float*)d_in[1];
    const float* bq = (const float*)d_in[2];
    const float* Wk = (const float*)d_in[3];
    const float* bk = (const float*)d_in[4];
    const float* Wv = (const float*)d_in[5];
    const float* bv = (const float*)d_in[6];
    const float* We = (const float*)d_in[7];
    const float* be = (const float*)d_in[8];
    const float* Wo = (const float*)d_in[9];
    const float* bo = (const float*)d_in[10];
    float* out = (float*)d_out;

    char* ws = (char*)d_ws;
    size_t off = 0;
    auto alloc = [&](size_t bytes) -> void* {
        void* p = ws + off;
        off = (off + bytes + 255) & ~(size_t)255;
        return p;
    };
    unsigned short* xb   = (unsigned short*)alloc((size_t)ROWS * HID * 2);
    unsigned short* vbT  = (unsigned short*)alloc((size_t)ROWS * HID * 2);  // [b][HID][SEQ]
    unsigned short* WvbT = (unsigned short*)alloc((size_t)HID * HID * 2);   // [N][K]
    unsigned short* WobT = (unsigned short*)alloc((size_t)HID * HID * 2);   // [N][K]
    float* Wqe  = (float*)alloc((size_t)HID * QDIM * 4);
    float* Wke  = (float*)alloc((size_t)HID * QDIM * 4);
    float* bqe  = (float*)alloc(256);
    float* bke  = (float*)alloc(256);
    float* qq   = (float*)alloc((size_t)ROWS * QDIM * 4);
    float* kq   = (float*)alloc((size_t)ROWS * QDIM * 4);
    float* rsin = (float*)alloc((size_t)ROWS * 4);
    unsigned short* P  = (unsigned short*)alloc((size_t)BATCH * SEQ * SEQ * 2);
    unsigned short* ob = (unsigned short*)alloc((size_t)ROWS * HID * 2);
    (void)ws_size; (void)n_in; (void)in_sizes; (void)out_size;

    // 1) precision conversion (weights transposed for async-friendly staging)
    cvt_bf16_kernel<<<2048, 256, 0, stream>>>(x, xb, ROWS * HID);
    cvt_bf16_T_kernel<<<dim3(HID / 32, HID / 32), 256, 0, stream>>>(Wv, WvbT, HID, HID);
    cvt_bf16_T_kernel<<<dim3(HID / 32, HID / 32), 256, 0, stream>>>(Wo, WobT, HID, HID);

    // 2) fold We through Wq/Wk (kills two full GEMMs)
    folded_we_kernel<<<(2 * HID * QDIM + 255) / 256, 256, 0, stream>>>(
        Wq, Wk, We, Wqe, Wke);
    folded_bias_kernel<<<1, 32, 0, stream>>>(bq, bk, We, be, bqe, bke);

    // 3) quantum embeds, unit-normalized
    embed_norm_kernel<<<ROWS / 16, 256, 0, stream>>>(x, Wqe, Wke, bqe, bke, qq, kq);

    // 4) V projection, written TRANSPOSED per batch: vbT[b] = (x_b@Wv+bv)^T
    gemm_bf16_kernel<false, true, false, true>
        <<<dim3(HID / 64, SEQ / 128, BATCH), 256, 0, stream>>>(
        xb, WvbT, bv, nullptr, vbT, SEQ, HID, HID,
        (long)SEQ * HID, 0, (long)HID * SEQ, 0);

    // 5) fidelity scores P = exp(<q|k>^2), row sums
    scores_kernel<<<dim3(SEQ, BATCH), 256, 0, stream>>>(qq, kq, P, rsin);

    // 6) attended = diag(1/rowsum) * (P @ v)   (B = vbT, already [N][K])
    gemm_bf16_kernel<false, false, true, false>
        <<<dim3(HID / 64, SEQ / 128, BATCH), 256, 0, stream>>>(
        P, vbT, nullptr, rsin, ob, SEQ, HID, SEQ,
        (long)SEQ * SEQ, (long)HID * SEQ, (long)SEQ * HID, SEQ);

    // 7) out = attended @ Wo + bo (fp32)
    gemm_bf16_kernel<true, true, false, false>
        <<<dim3(HID / 64, ROWS / 128, 1), 256, 0, stream>>>(
        ob, WobT, bo, nullptr, out, ROWS, HID, HID, 0, 0, 0, 0);
}
